// Create_recurrent_unit_39101382263299
// MI455X (gfx1250) — matrix operand order repaired and checked by reading, not compiled, not run on hardware
//
#include <hip/hip_runtime.h>
#include <hip/hip_bf16.h>

typedef __attribute__((ext_vector_type(16))) __bf16 bf16x16;
typedef __attribute__((ext_vector_type(8)))  float  f32x8;

constexpr int Bv = 4096;   // batch
constexpr int Ev = 1024;   // embed
constexpr int Hv = 1024;   // hidden
constexpr int K2 = 2048;   // E + H (fused K dimension)

__device__ __forceinline__ unsigned short f2bf(float f) {
  unsigned u = __float_as_uint(f);
  u += 0x7FFFu + ((u >> 16) & 1u);          // round-to-nearest-even
  return (unsigned short)(u >> 16);
}

// ---------------------------------------------------------------------------
// Kernel 1: pack [x | h_prev] (fp32) -> A_bf [B][2048] bf16
// ---------------------------------------------------------------------------
__global__ __launch_bounds__(256) void pack_A(const float* __restrict__ x,
                                              const float* __restrict__ h,
                                              unsigned short* __restrict__ A_bf) {
  int idx4 = blockIdx.x * 256 + threadIdx.x;   // one float4 of x and of h each
  int row  = idx4 >> 8;                        // / (1024/4)
  int c4   = (idx4 & 255) << 2;                // column base
  float4 vx = *(const float4*)(x + (size_t)row * Ev + c4);
  float4 vh = *(const float4*)(h + (size_t)row * Hv + c4);
  uint2 px, ph;
  px.x = (unsigned)f2bf(vx.x) | ((unsigned)f2bf(vx.y) << 16);
  px.y = (unsigned)f2bf(vx.z) | ((unsigned)f2bf(vx.w) << 16);
  ph.x = (unsigned)f2bf(vh.x) | ((unsigned)f2bf(vh.y) << 16);
  ph.y = (unsigned)f2bf(vh.z) | ((unsigned)f2bf(vh.w) << 16);
  *(uint2*)(A_bf + (size_t)row * K2 + c4)       = px;
  *(uint2*)(A_bf + (size_t)row * K2 + Ev + c4)  = ph;
}

// ---------------------------------------------------------------------------
// Kernel 2: transpose + convert weights -> Wt[g][n][k] bf16, k in [0,2048)
//           k < 1024 comes from W_g[k][n], else U_g[k-1024][n]
// ---------------------------------------------------------------------------
__global__ __launch_bounds__(256) void pack_Wt(
    const float* __restrict__ Wi, const float* __restrict__ Ui,
    const float* __restrict__ Wf, const float* __restrict__ Uf,
    const float* __restrict__ Wo, const float* __restrict__ Uo,
    const float* __restrict__ Wc, const float* __restrict__ Uc,
    unsigned short* __restrict__ Wt) {
  int g = blockIdx.z;
  const float* Wg = (g == 0) ? Wi : (g == 1) ? Wf : (g == 2) ? Wo : Wc;
  const float* Ug = (g == 0) ? Ui : (g == 1) ? Uf : (g == 2) ? Uo : Uc;

  __shared__ float tile[32][33];
  int tx = threadIdx.x & 31;
  int ty = threadIdx.x >> 5;
  int k0 = blockIdx.x * 32;
  int n0 = blockIdx.y * 32;

#pragma unroll
  for (int i = 0; i < 4; ++i) {
    int k = k0 + ty + i * 8;
    const float* src = (k < Ev) ? (Wg + (size_t)k * Hv) : (Ug + (size_t)(k - Ev) * Hv);
    tile[ty + i * 8][tx] = src[n0 + tx];
  }
  __syncthreads();
#pragma unroll
  for (int i = 0; i < 4; ++i) {
    int n = n0 + ty + i * 8;
    Wt[((size_t)g * Hv + n) * K2 + k0 + tx] = f2bf(tile[tx][ty + i * 8]);
  }
}

// ---------------------------------------------------------------------------
// Async global -> LDS DMA (CDNA5, ASYNCcnt-tracked), GVS addressing:
//   mem = SADDR(base) + VADDR(32-bit byte offset); VDST = LDS byte address.
//   ldsAddr comes from ptrtoint of the real shared object so the LDS global
//   escapes into the asm (keeps AA honest about the asm writing LDS).
// ---------------------------------------------------------------------------
__device__ __forceinline__ void async_b128(unsigned ldsAddr, unsigned gOff,
                                           unsigned long long base) {
  asm volatile("global_load_async_to_lds_b128 %0, %1, %2"
               :: "v"(ldsAddr), "v"(gOff), "s"(base)
               : "memory");
}

// ---------------------------------------------------------------------------
// Kernel 3: fused GEMM (bf16 WMMA, fp32 acc) + LSTM pointwise epilogue
//   block tile: M=256, N=32; 8 waves, each wave = 32 rows x 32 cols x 4 gates
//   double-buffered LDS filled by async DMA, two barriers per K-step
// ---------------------------------------------------------------------------
struct alignas(16) Smem {
  __bf16 A[2][256][40];    // 2 x 20480 B  (+8 halves pad -> 80B row stride)
  __bf16 B[2][4][32][40];  // 2 x 10240 B  [buf][gate][n][k]
};
constexpr unsigned SM_A_BUF = 256 * 40 * 2;            // 20480: A buffer stride
constexpr unsigned SM_B_BUF = 4 * 32 * 40 * 2;         // 10240: B buffer stride

__global__ __launch_bounds__(256) void lstm_gemm(
    const unsigned short* __restrict__ A_bf,   // [B][2048] bf16
    const unsigned short* __restrict__ Wt,     // [4][1024][2048] bf16 (n-major)
    const float* __restrict__ c_prev,          // [B][H]
    const float* __restrict__ pbi, const float* __restrict__ pbf,
    const float* __restrict__ pbo, const float* __restrict__ pbc,
    float* __restrict__ out)                   // [2][B][H]
{
  __shared__ Smem sm;

  const int tid  = threadIdx.x;
  const int lane = tid & 31;
  const int wv   = tid >> 5;
  const int m0   = blockIdx.y * 256;
  const int n0   = blockIdx.x * 32;
  const int laneLo = lane & 15;
  const int laneHi = lane >> 4;   // 0 or 1

  f32x8 acc[4][2][2];             // [gate][mSub][nSub] = 128 VGPRs
  const f32x8 zero = {0.f, 0.f, 0.f, 0.f, 0.f, 0.f, 0.f, 0.f};
#pragma unroll
  for (int g = 0; g < 4; ++g)
#pragma unroll
    for (int ms = 0; ms < 2; ++ms)
#pragma unroll
      for (int t = 0; t < 2; ++t) acc[g][ms][t] = zero;

  // --- per-thread DMA slot addresses (constant across K loop) ---
  // LDS addresses are ptrtoint of the actual shared object (low 32 bits of a
  // generic LDS pointer are the LDS byte address per the aperture mapping).
  unsigned ldsA[4], gA[4];
#pragma unroll
  for (int i = 0; i < 4; ++i) {
    int s = tid + i * 256;                   // 1024 slots: A tile 256x32
    int r = s >> 2, seg = s & 3;
    ldsA[i] = (unsigned)(uintptr_t)&sm.A[0][r][seg * 8];
    gA[i]   = (unsigned)(((m0 + r) * K2 + seg * 8) * 2);
  }
  unsigned ldsB[2], gB[2];
#pragma unroll
  for (int i = 0; i < 2; ++i) {
    int s = tid + i * 256;                   // 512 slots: 4 x (32x32) B tiles
    int g = s >> 7, rem = s & 127, n = rem >> 2, seg = rem & 3;
    ldsB[i] = (unsigned)(uintptr_t)&sm.B[0][g][n][seg * 8];
    gB[i]   = (unsigned)(((g * Hv + n0 + n) * K2 + seg * 8) * 2);
  }
  const unsigned long long aBase = (unsigned long long)A_bf;
  const unsigned long long bBase = (unsigned long long)Wt;

  auto issue = [&](int buf, int kk) {
    const unsigned aOff = (unsigned)buf * SM_A_BUF;
    const unsigned bOff = (unsigned)buf * SM_B_BUF;
    const unsigned kByte = (unsigned)(kk * 2);
#pragma unroll
    for (int i = 0; i < 4; ++i) async_b128(ldsA[i] + aOff, gA[i] + kByte, aBase);
#pragma unroll
    for (int i = 0; i < 2; ++i) async_b128(ldsB[i] + bOff, gB[i] + kByte, bBase);
  };

  issue(0, 0);                                // prologue: stage 0

  for (int it = 0; it < K2 / 32; ++it) {
    const int buf = it & 1;
    asm volatile("s_wait_asynccnt 0x0" ::: "memory");  // stage `it` landed
    __syncthreads();                          // ...for every wave; buf^1 free
    if (it + 1 < K2 / 32) issue(buf ^ 1, (it + 1) * 32);  // DMA overlaps WMMA

    // ---- A fragments (ISA 16-bit A 16x32 layout), 2 M-subtiles ----
    bf16x16 aF[2];
#pragma unroll
    for (int ms = 0; ms < 2; ++ms) {
      const __bf16* ap = &sm.A[buf][wv * 32 + ms * 16 + laneLo][laneHi * 8];
      ((uint4*)&aF[ms])[0] = *(const uint4*)ap;         // K 0..7  (or 8..15)
      ((uint4*)&aF[ms])[1] = *(const uint4*)(ap + 16);  // K 16..23 (or 24..31)
    }
    // ---- 4 gates x 2 N-subtiles, each B fragment reused for 2 M-subtiles ----
#pragma unroll
    for (int g = 0; g < 4; ++g) {
#pragma unroll
      for (int t = 0; t < 2; ++t) {
        bf16x16 bF;
        const __bf16* bp = &sm.B[buf][g][t * 16 + laneLo][laneHi * 8];
        ((uint4*)&bF)[0] = *(const uint4*)bp;           // K 0..7 (or 8..15)
        ((uint4*)&bF)[1] = *(const uint4*)(bp + 16);    // K 16..23 (or 24..31)
#pragma unroll
        for (int ms = 0; ms < 2; ++ms)
          acc[g][ms][t] = __builtin_amdgcn_wmma_f32_16x16x32_bf16(
              false, aF[ms], false, bF, (short)0, acc[g][ms][t], false, false);
      }
    }
    __syncthreads();                          // done reading buf before refill
  }

  // ---- fused LSTM epilogue, in-register ----
  const int baseRowW = m0 + wv * 32 + (laneHi ? 8 : 0);
#pragma unroll
  for (int ms = 0; ms < 2; ++ms) {
#pragma unroll
    for (int t = 0; t < 2; ++t) {
      const int col = n0 + t * 16 + laneLo;
      const float bI = pbi[col], bF_ = pbf[col], bO = pbo[col], bC = pbc[col];
#pragma unroll
      for (int r = 0; r < 8; ++r) {
        const int row = baseRowW + ms * 16 + r;
        const size_t off = (size_t)row * Hv + col;
        const float iv = 1.0f / (1.0f + __expf(-(acc[0][ms][t][r] + bI)));
        const float fv = 1.0f / (1.0f + __expf(-(acc[1][ms][t][r] + bF_)));
        const float ov = 1.0f / (1.0f + __expf(-(acc[2][ms][t][r] + bO)));
        const float gx = acc[3][ms][t][r] + bC;
        const float cg = 1.0f - 2.0f / (1.0f + __expf(2.0f * gx));    // tanh
        const float c  = fv * c_prev[off] + iv * cg;
        const float th = 1.0f - 2.0f / (1.0f + __expf(2.0f * c));     // tanh
        out[off]                   = ov * th;   // h
        out[(size_t)Bv * Hv + off] = c;         // c
      }
    }
  }
}

// ---------------------------------------------------------------------------
extern "C" void kernel_launch(void* const* d_in, const int* in_sizes, int n_in,
                              void* d_out, int out_size, void* d_ws, size_t ws_size,
                              hipStream_t stream) {
  const float* x   = (const float*)d_in[0];
  const float* hm  = (const float*)d_in[1];   // [2][B][H]: h_prev, c_prev
  const float* Wi  = (const float*)d_in[2];
  const float* Ui  = (const float*)d_in[3];
  const float* bi  = (const float*)d_in[4];
  const float* Wf  = (const float*)d_in[5];
  const float* Uf  = (const float*)d_in[6];
  const float* bfp = (const float*)d_in[7];
  const float* Wo  = (const float*)d_in[8];
  const float* Uo  = (const float*)d_in[9];
  const float* bo  = (const float*)d_in[10];
  const float* Wc  = (const float*)d_in[11];
  const float* Uc  = (const float*)d_in[12];
  const float* bc  = (const float*)d_in[13];
  float* out = (float*)d_out;

  const float* h_prev = hm;
  const float* c_prev = hm + (size_t)Bv * Hv;

  unsigned short* A_bf = (unsigned short*)d_ws;            // 16 MB
  unsigned short* Wt   = A_bf + (size_t)Bv * K2;           // 16 MB

  pack_A<<<(Bv * (Ev / 4)) / 256, 256, 0, stream>>>(x, h_prev, A_bf);
  pack_Wt<<<dim3(K2 / 32, Hv / 32, 4), 256, 0, stream>>>(Wi, Ui, Wf, Uf, Wo, Uo,
                                                         Wc, Uc, Wt);
  lstm_gemm<<<dim3(Hv / 32, Bv / 256), 256, 0, stream>>>(A_bf, Wt, c_prev,
                                                         bi, bfp, bo, bc, out);
}
